// SelfAttentionModule_65481071403601
// MI455X (gfx1250) — matrix operand order repaired and checked by reading, not compiled, not run on hardware
//
#include <hip/hip_runtime.h>
#include <cmath>

typedef __bf16 bf16_t;
typedef __attribute__((ext_vector_type(16))) __bf16 v16bf;
typedef __attribute__((ext_vector_type(8)))  float  v8f;
typedef int v4i __attribute__((__vector_size__(4 * sizeof(int))));

constexpr int Bb = 4, Ss = 2048, Ee = 512;
constexpr int Nn = Bb * Ss;           // 8192 rows

// ---- gfx1250 async-to-LDS path (guarded so host pass / older clang still compiles)
#if defined(__has_builtin)
#if __has_builtin(__builtin_amdgcn_global_load_async_to_lds_b128)
#define HAVE_ASYNC_LDS 1
#endif
#if __has_builtin(__builtin_amdgcn_s_wait_asynccnt)
#define HAVE_WAIT_ASYNC 1
#endif
#endif

__device__ __forceinline__ void async_cp16(bf16_t* dst_lds, const bf16_t* src_g) {
#ifdef HAVE_ASYNC_LDS
  typedef __attribute__((address_space(1))) v4i* gv4i_p;   // global (__device__)
  typedef __attribute__((address_space(3))) v4i* lv4i_p;   // LDS (__shared__)
  __builtin_amdgcn_global_load_async_to_lds_b128(
      (gv4i_p)(void*)src_g, (lv4i_p)(void*)dst_lds, 0, 0);
#else
  *(uint4*)dst_lds = *(const uint4*)src_g;
#endif
}

__device__ __forceinline__ void async_drain() {
#ifdef HAVE_WAIT_ASYNC
  __builtin_amdgcn_s_wait_asynccnt(0);
#elif defined(HAVE_ASYNC_LDS)
  asm volatile("s_wait_asynccnt 0" ::: "memory");
#endif
}

// ---------------------------------------------------------------------------
// Projection GEMM: C_bf16[N,E] = bf16(A_f32[N,E]) @ bf16(W_f32[E,E])
// One 16x16 output tile per wave, K-loop step 32, v_wmma_f32_16x16x32_bf16.
// ---------------------------------------------------------------------------
__global__ __launch_bounds__(256) void proj_gemm(const float* __restrict__ A,
                                                 const float* __restrict__ W,
                                                 bf16_t* __restrict__ C) {
  const int lane    = threadIdx.x & 31;
  const int wave    = threadIdx.x >> 5;
  const int l15     = lane & 15;
  const int halfsel = lane >> 4;                // 0: lanes 0-15, 1: lanes 16-31
  const int tile    = blockIdx.x * 8 + wave;
  const int tilesN  = Ee / 16;                  // 32
  const int tm      = tile / tilesN;
  const int tn      = tile % tilesN;

  v8f acc = {};
  for (int k0 = 0; k0 < Ee; k0 += 32) {
    v16bf a, b;
    // A operand: lane row = tm*16 + l15; K halves: {0..7,16..23} + halfsel*8
    const float* arow = A + (size_t)(tm * 16 + l15) * Ee + k0 + halfsel * 8;
#pragma unroll
    for (int h = 0; h < 16; ++h) {
      int kk = (h < 8) ? h : (h + 8);
      a[h] = (bf16_t)arow[kk];
    }
    // B operand: lane col = tn*16 + l15; K halves: {0..7,16..23} + halfsel*8 (stride E rows)
    const float* bcol = W + (size_t)(k0 + halfsel * 8) * Ee + tn * 16 + l15;
#pragma unroll
    for (int h = 0; h < 16; ++h) b[h] = (bf16_t)bcol[(size_t)((h < 8) ? h : (h + 8)) * Ee];

    acc = __builtin_amdgcn_wmma_f32_16x16x32_bf16(false, a, false, b,
                                                  (short)0, acc, false, false);
  }
#pragma unroll
  for (int r = 0; r < 8; ++r) {
    int row = tm * 16 + r + halfsel * 8;
    int col = tn * 16 + l15;
    C[(size_t)row * Ee + col] = (bf16_t)acc[r];
  }
}

// ---------------------------------------------------------------------------
// X (f32, [N,E] row-major) -> X^T (bf16, [E,N]) so P@X B-operands load
// contiguously (consecutive keys per lane).
// ---------------------------------------------------------------------------
__global__ void xpose_bf16(const float* __restrict__ X, bf16_t* __restrict__ XT) {
  size_t i = (size_t)blockIdx.x * blockDim.x + threadIdx.x;
  if (i >= (size_t)Nn * Ee) return;
  size_t n = i / Ee, e = i % Ee;
  XT[e * (size_t)Nn + n] = (bf16_t)X[i];
}

// ---------------------------------------------------------------------------
// Flash attention: block = 32 query rows (2 slabs of 16), 8 waves split E
// into 64-col slices. Per 32-key step: K tile async-DMA'd into LDS once per
// block and its B-operands reused by BOTH slabs (2 WMMAs per operand load),
// partial QK^T via WMMA + ds_add_f32 reduce, online softmax per slab,
// rescale, P@X via WMMA (bX loaded once, fed to both slabs).
// Logits never touch HBM; K/X^T (16 MB) stay L2-resident across blocks.
// ---------------------------------------------------------------------------
__global__ __launch_bounds__(256) void flash_attn(const bf16_t* __restrict__ Qb,
                                                  const bf16_t* __restrict__ Kb,
                                                  const bf16_t* __restrict__ XbT,
                                                  float* __restrict__ Out) {
  __shared__ bf16_t Kt[32 * Ee];        // 32 KB staged K tile
  __shared__ float  S_t[2][16 * 32];    // per-slab score tiles
  __shared__ bf16_t P_t[2][16 * 32];
  __shared__ float  m_row[32], l_row[32], c_row[32];

  const int tid     = threadIdx.x;
  const int lane    = tid & 31;
  const int wave    = tid >> 5;
  const int l15     = lane & 15;
  const int halfsel = lane >> 4;
  const int row0    = blockIdx.x * 32;   // 32 query rows per block
  const int e0      = wave * 64;         // this wave's E slice
  const float scale = 0.044194173824159216f; // 1/sqrt(512)

  if (tid < 32) { m_row[tid] = -INFINITY; l_row[tid] = 0.f; c_row[tid] = 0.f; }

  // Preload Q A-operands: 2 slabs x 2 K-chunks of 32 for this wave's E slice
  v16bf aQ[2][2];
#pragma unroll
  for (int s = 0; s < 2; ++s)
#pragma unroll
    for (int c = 0; c < 2; ++c) {
      const bf16_t* qrow =
          Qb + (size_t)(row0 + s * 16 + l15) * Ee + e0 + c * 32 + halfsel * 16;
#pragma unroll
      for (int h = 0; h < 16; ++h) {
        int kk = h;
        aQ[s][c][h] = qrow[kk];
      }
    }

  v8f oacc[2][4] = {{{}, {}, {}, {}}, {{}, {}, {}, {}}}; // 2 x (16x64) f32
  __syncthreads();

  for (int kb = 0; kb < Nn; kb += 32) {
    // ---- async-stage K tile (32 rows x 512 bf16): 2048 x b128, 8 per thread
#pragma unroll
    for (int i = 0; i < 8; ++i) {
      int idx   = tid + i * 256;         // 0..2047
      int krow  = idx >> 6;              // 64 x 16B chunks per row
      int chunk = idx & 63;
      async_cp16(&Kt[krow * Ee + chunk * 8],
                 Kb + (size_t)(kb + krow) * Ee + chunk * 8);
    }
    // ---- prefetch next K tile while this one is in flight
    if (kb + 32 < Nn) {
      int prow = tid >> 3, poff = (tid & 7) * 64;
      __builtin_prefetch(Kb + (size_t)(kb + 32 + prow) * Ee + poff, 0, 1);
    }
    // ---- zero score tiles (2 x 512 floats)
    for (int i = tid; i < 2 * 16 * 32; i += 256) (&S_t[0][0])[i] = 0.f;
    async_drain();
    __syncthreads();

    // ---- partial S = Q K^T: load each K B-operand once, use for both slabs
    v16bf bK[2][2];
#pragma unroll
    for (int kt = 0; kt < 2; ++kt)
#pragma unroll
      for (int c = 0; c < 2; ++c)
        bK[kt][c] = *(const v16bf*)&Kt[(kt * 16 + l15) * Ee + e0 + c * 32 + halfsel * 16];

#pragma unroll
    for (int s = 0; s < 2; ++s)
#pragma unroll
      for (int kt = 0; kt < 2; ++kt) {
        v8f sacc = {};
#pragma unroll
        for (int c = 0; c < 2; ++c)
          sacc = __builtin_amdgcn_wmma_f32_16x16x32_bf16(false, aQ[s][c], false, bK[kt][c],
                                                         (short)0, sacc, false, false);
#pragma unroll
        for (int r = 0; r < 8; ++r) {
          int m = r + halfsel * 8;
          int n = kt * 16 + l15;
          atomicAdd(&S_t[s][m * 32 + n], sacc[r]);
        }
      }
    __syncthreads();

    // ---- online softmax per slab: thread -> (m = tid/16, cols n, n+16)
#pragma unroll
    for (int s = 0; s < 2; ++s) {
      int m = tid >> 4, nn = tid & 15;
      float s0 = S_t[s][m * 32 + nn] * scale;
      float s1 = S_t[s][m * 32 + nn + 16] * scale;
      float mx = fmaxf(s0, s1);
#pragma unroll
      for (int off = 8; off >= 1; off >>= 1)
        mx = fmaxf(mx, __shfl_xor(mx, off, 16));
      float m_old = m_row[s * 16 + m];
      float m_new = fmaxf(m_old, mx);
      float p0 = __expf(s0 - m_new);
      float p1 = __expf(s1 - m_new);
      float sum = p0 + p1;
#pragma unroll
      for (int off = 8; off >= 1; off >>= 1)
        sum += __shfl_xor(sum, off, 16);
      P_t[s][m * 32 + nn]      = (bf16_t)p0;
      P_t[s][m * 32 + nn + 16] = (bf16_t)p1;
      if (nn == 0) {
        float corr = __expf(m_old - m_new);
        c_row[s * 16 + m] = corr;
        l_row[s * 16 + m] = l_row[s * 16 + m] * corr + sum;
        m_row[s * 16 + m] = m_new;
      }
    }
    __syncthreads();

    // ---- rescale accumulators
#pragma unroll
    for (int s = 0; s < 2; ++s) {
      float cv[8];
#pragma unroll
      for (int r = 0; r < 8; ++r) cv[r] = c_row[s * 16 + r + halfsel * 8];
#pragma unroll
      for (int nt = 0; nt < 4; ++nt)
#pragma unroll
        for (int r = 0; r < 8; ++r) oacc[s][nt][r] *= cv[r];
    }

    // ---- O += P @ X: load each bX once, feed both slabs
    v16bf aP[2];
#pragma unroll
    for (int s = 0; s < 2; ++s)
#pragma unroll
      for (int h = 0; h < 16; ++h) {
        int kk = h;
        aP[s][h] = P_t[s][l15 * 32 + halfsel * 16 + kk];
      }
#pragma unroll
    for (int nt = 0; nt < 4; ++nt) {
      int ecol = e0 + nt * 16 + l15;                 // B col = embed column
      v16bf bX = *(const v16bf*)(XbT + (size_t)ecol * Nn + kb + halfsel * 16);
#pragma unroll
      for (int s = 0; s < 2; ++s)
        oacc[s][nt] = __builtin_amdgcn_wmma_f32_16x16x32_bf16(false, aP[s], false, bX,
                                                              (short)0, oacc[s][nt],
                                                              false, false);
    }
    __syncthreads();
  }

  // ---- finalize: divide by row sums, store f32
#pragma unroll
  for (int s = 0; s < 2; ++s) {
    float invl[8];
#pragma unroll
    for (int r = 0; r < 8; ++r) invl[r] = 1.0f / l_row[s * 16 + r + halfsel * 8];
#pragma unroll
    for (int nt = 0; nt < 4; ++nt)
#pragma unroll
      for (int r = 0; r < 8; ++r) {
        int m = r + halfsel * 8;
        int n = e0 + nt * 16 + l15;
        Out[(size_t)(row0 + s * 16 + m) * Ee + n] = oacc[s][nt][r] * invl[r];
      }
  }
}

// ---------------------------------------------------------------------------
extern "C" void kernel_launch(void* const* d_in, const int* in_sizes, int n_in,
                              void* d_out, int out_size, void* d_ws, size_t ws_size,
                              hipStream_t stream) {
  const float* x   = (const float*)d_in[0];
  const float* rot = (const float*)d_in[1];
  const float* ent = (const float*)d_in[2];
  float* out = (float*)d_out;

  bf16_t* Qb  = (bf16_t*)d_ws;                 //  8 MiB
  bf16_t* Kb  = Qb  + (size_t)Nn * Ee;         //  8 MiB
  bf16_t* XbT = Kb  + (size_t)Nn * Ee;         //  8 MiB

  const int tiles = (Nn / 16) * (Ee / 16);     // 16384 tiles, 8 per block
  proj_gemm<<<tiles / 8, 256, 0, stream>>>(x, rot, Qb);
  proj_gemm<<<tiles / 8, 256, 0, stream>>>(x, ent, Kb);
  xpose_bf16<<<(Nn * Ee + 255) / 256, 256, 0, stream>>>(x, XbT);
  flash_attn<<<Nn / 32, 256, 0, stream>>>(Qb, Kb, XbT, out);
}
